// FixedEdgeWeightGNN_38878043964035
// MI455X (gfx1250) — matrix operand order repaired and checked by reading, not compiled, not run on hardware
//
#include <hip/hip_runtime.h>

#define NN   50000
#define EE   600000
#define DIM  128
#define TEMB 8
#define INR  392          // real e_in width
#define INP  416          // padded to 13 * 32 for WMMA K-steps

typedef __bf16 v16bf __attribute__((ext_vector_type(16)));
typedef __bf16 v8bf  __attribute__((ext_vector_type(8)));
typedef float  v8f   __attribute__((ext_vector_type(8)));

__device__ __forceinline__ unsigned short f2bf_bits(float f) {
  unsigned u = __float_as_uint(f);
  unsigned r = u + 0x7FFFu + ((u >> 16) & 1u);   // round-to-nearest-even
  return (unsigned short)(r >> 16);
}

__device__ __forceinline__ unsigned pack2bf(float lo, float hi) {
  return (unsigned)f2bf_bits(lo) | ((unsigned)f2bf_bits(hi) << 16);
}

// B fragment (32x16 KxN, bf16): lane = col(lane&15), K-half (lane>>4)*16.
// B[k][n] == W[n][k] for x @ W.T, so load straight from row-major weight rows.
__device__ __forceinline__ v16bf load_b_frag(const unsigned short* __restrict__ W,
                                             int nbase, int kbase, int ldk) {
  int lane = threadIdx.x & 31;
  const unsigned short* p =
      W + (size_t)(nbase + (lane & 15)) * ldk + kbase + ((lane >> 4) << 4);
  return *(const v16bf*)p;
}

// A fragment (16x32 MxK, bf16) from row-major LDS tile: row = lane&15,
// chunks at kbase + (lane>>4)*16 and +8 (contiguous K-half, matches B).
__device__ __forceinline__ v16bf load_a_frag_lds(const unsigned short* A,
                                                 int kbase, int ld) {
  int lane = threadIdx.x & 31;
  const unsigned short* row = A + (lane & 15) * ld;
  int k0 = kbase + ((lane >> 4) << 4);
  v8bf c0 = *(const v8bf*)(row + k0);
  v8bf c1 = *(const v8bf*)(row + k0 + 8);
  return __builtin_shufflevector(c0, c1, 0, 1, 2, 3, 4, 5, 6, 7,
                                 8, 9, 10, 11, 12, 13, 14, 15);
}

// ---------------- prep: convert weights to bf16 (ew1 zero-padded to 416) ----
__global__ void prep_weights_kernel(const float* __restrict__ ew1,
                                    const float* __restrict__ mw1,
                                    const float* __restrict__ mw2,
                                    unsigned short* __restrict__ ew1bf,
                                    unsigned short* __restrict__ mw1bf,
                                    unsigned short* __restrict__ mw2bf) {
  int i = blockIdx.x * blockDim.x + threadIdx.x;
  if (i < DIM * INP) {
    int n = i / INP, k = i % INP;
    ew1bf[i] = (k < INR) ? f2bf_bits(ew1[n * INR + k]) : (unsigned short)0;
  }
  if (i < DIM * DIM) {
    mw1bf[i] = f2bf_bits(mw1[i]);
    mw2bf[i] = f2bf_bits(mw2[i]);
  }
}

__global__ void init_kernel(float* __restrict__ agg, float* __restrict__ maxbuf,
                            float* __restrict__ sumbuf) {
  int i = blockIdx.x * blockDim.x + threadIdx.x;
  if (i < NN * DIM) agg[i] = 0.f;
  if (i < NN) { maxbuf[i] = 0.f; sumbuf[i] = 0.f; }
}

// ---------------- edge MLP: w[e] = sigmoid(relu(e_in @ ew1.T + eb1) . ew2) ---
__global__ __launch_bounds__(256) void edge_mlp_kernel(
    const float* __restrict__ z, const int* __restrict__ src,
    const int* __restrict__ dst, const int* __restrict__ etype,
    const float* __restrict__ type_emb, const unsigned short* __restrict__ ew1bf,
    const float* __restrict__ eb1, const float* __restrict__ ew2,
    const float* __restrict__ eb2, float* __restrict__ wbuf,
    unsigned int* __restrict__ maxbuf, int numTiles) {
  __shared__ __align__(16) unsigned short sA[16 * INP];  // e_in tile, bf16
  __shared__ float sH[16 * DIM];                         // hidden activations
  __shared__ int sSrc[16], sDst[16], sEt[16];

  const int tid  = threadIdx.x;
  const int lane = tid & 31;
  const int wave = tid >> 5;         // 8 waves: wave w owns cols [16w,16w+16)
  const int nbase = wave << 4;
  const int col  = lane & 15;
  const int half = lane >> 4;

  v16bf bfrag[13];                   // preload B once per persistent block
#pragma unroll
  for (int kk = 0; kk < 13; ++kk)
    bfrag[kk] = load_b_frag(ew1bf, nbase, kk * 32, INP);
  const float bias = eb1[nbase + col];

  for (int tile = blockIdx.x; tile < numTiles; tile += gridDim.x) {
    __syncthreads();
    if (tid < 16) {
      int e = tile * 16 + tid;
      sSrc[tid] = src[e]; sDst[tid] = dst[e]; sEt[tid] = etype[e];
    }
    __syncthreads();
    // build bf16 e_in = [zi | zj | |zi-zj| | te | 0-pad], two elements at a time
    for (int idx = tid; idx < 16 * (DIM / 2); idx += 256) {
      int e = idx >> 6, d2 = (idx & 63) << 1;
      const float2 a2 = *(const float2*)(z + (size_t)sSrc[e] * DIM + d2);
      const float2 b2 = *(const float2*)(z + (size_t)sDst[e] * DIM + d2);
      unsigned* row32 = (unsigned*)sA + e * (INP / 2);
      row32[d2 >> 1]             = pack2bf(a2.x, a2.y);
      row32[(DIM + d2) >> 1]     = pack2bf(b2.x, b2.y);
      row32[(2 * DIM + d2) >> 1] =
          pack2bf(fabsf(a2.x - b2.x), fabsf(a2.y - b2.y));
    }
    for (int idx = tid; idx < 16 * 16; idx += 256) {   // k pairs 384..415
      int e = idx >> 4, k2 = 3 * DIM + ((idx & 15) << 1);
      unsigned v = 0;
      if (k2 < INR) {
        const float* te = type_emb + sEt[e] * TEMB + (k2 - 3 * DIM);
        v = pack2bf(te[0], te[1]);
      }
      ((unsigned*)sA)[e * (INP / 2) + (k2 >> 1)] = v;
    }
    __syncthreads();

    // preload the full A tile into registers (all LDS loads in flight at once),
    // then issue the 13-deep WMMA accumulation chain back-to-back
    v16bf afrag[13];
#pragma unroll
    for (int kk = 0; kk < 13; ++kk)
      afrag[kk] = load_a_frag_lds(sA, kk * 32, INP);
    v8f c = {};
#pragma unroll
    for (int kk = 0; kk < 13; ++kk)
      c = __builtin_amdgcn_wmma_f32_16x16x32_bf16(false, afrag[kk], false,
                                                  bfrag[kk], (short)0, c,
                                                  false, false);
#pragma unroll
    for (int v = 0; v < 8; ++v) {
      int m = v + (half << 3);
      float h = c[v] + bias;
      sH[m * DIM + nbase + col] = h > 0.f ? h : 0.f;
    }
    __syncthreads();

    // dot with ew2 + sigmoid; wave w reduces edges 2w and 2w+1
#pragma unroll
    for (int j = 0; j < 2; ++j) {
      int m = wave * 2 + j;
      float p = 0.f;
#pragma unroll
      for (int i = 0; i < 4; ++i) {
        int d = lane + 32 * i;
        p += sH[m * DIM + d] * ew2[d];
      }
#pragma unroll
      for (int off = 16; off > 0; off >>= 1) p += __shfl_xor(p, off, 32);
      if (lane == 0) {
        float wsig = 1.f / (1.f + __expf(-(p + eb2[0])));
        wbuf[tile * 16 + m] = wsig;
        // wsig in (0,1): positive-float bit pattern is order-preserving as uint
        atomicMax(&maxbuf[sSrc[m]], __float_as_uint(wsig));
      }
    }
  }
}

// ---------------- segment softmax numerator + denominator --------------------
__global__ void softmax_exp_kernel(const float* __restrict__ wbuf,
                                   const int* __restrict__ src,
                                   const float* __restrict__ maxbuf,
                                   float* __restrict__ wexp,
                                   float* __restrict__ sumbuf, int e_count) {
  int e = blockIdx.x * blockDim.x + threadIdx.x;
  if (e >= e_count) return;
  int s = src[e];
  float we = __expf(wbuf[e] - maxbuf[s]);
  wexp[e] = we;
  atomicAdd(&sumbuf[s], we);
}

// ---------------- message MLP + alpha scale + scatter-add --------------------
__global__ __launch_bounds__(256) void message_kernel(
    const float* __restrict__ z, const int* __restrict__ src,
    const int* __restrict__ dst, const unsigned short* __restrict__ mw1bf,
    const float* __restrict__ mb1, const unsigned short* __restrict__ mw2bf,
    const float* __restrict__ mb2, const float* __restrict__ wexp,
    const float* __restrict__ sumbuf, float* __restrict__ alpha_out,
    float* __restrict__ agg, int numTiles) {
  __shared__ __align__(16) unsigned short sZ[16 * DIM];  // zj tile bf16
  __shared__ __align__(16) unsigned short sH[16 * DIM];  // hidden bf16
  __shared__ int sSrc[16], sDst[16];
  __shared__ float sAlpha[16];

  const int tid  = threadIdx.x;
  const int lane = tid & 31;
  const int wave = tid >> 5;
  const int nbase = wave << 4;
  const int col  = lane & 15;
  const int half = lane >> 4;

  v16bf b1[4], b2[4];
#pragma unroll
  for (int kk = 0; kk < 4; ++kk) {
    b1[kk] = load_b_frag(mw1bf, nbase, kk * 32, DIM);
    b2[kk] = load_b_frag(mw2bf, nbase, kk * 32, DIM);
  }
  const float bias1 = mb1[nbase + col];
  const float bias2 = mb2[nbase + col];

  for (int tile = blockIdx.x; tile < numTiles; tile += gridDim.x) {
    __syncthreads();
    if (tid < 16) {
      int e = tile * 16 + tid;
      int s = src[e];
      sSrc[tid] = s;
      sDst[tid] = dst[e];
      float a = wexp[e] / (sumbuf[s] + 1e-12f);
      sAlpha[tid] = a;
      alpha_out[e] = a;
    }
    __syncthreads();
    for (int idx = tid; idx < 16 * (DIM / 2); idx += 256) {
      int e = idx >> 6, d2 = (idx & 63) << 1;
      const float2 b2v = *(const float2*)(z + (size_t)sDst[e] * DIM + d2);
      ((unsigned*)sZ)[idx] = pack2bf(b2v.x, b2v.y);
    }
    __syncthreads();

    // stage 1: h = relu(zj @ mw1.T + mb1); preload A tile, then WMMA chain
    v16bf az[4];
#pragma unroll
    for (int kk = 0; kk < 4; ++kk) az[kk] = load_a_frag_lds(sZ, kk * 32, DIM);
    v8f c1 = {};
#pragma unroll
    for (int kk = 0; kk < 4; ++kk)
      c1 = __builtin_amdgcn_wmma_f32_16x16x32_bf16(false, az[kk], false, b1[kk],
                                                   (short)0, c1, false, false);
#pragma unroll
    for (int v = 0; v < 8; ++v) {
      int m = v + (half << 3);
      float h = c1[v] + bias1;
      sH[m * DIM + nbase + col] = f2bf_bits(h > 0.f ? h : 0.f);
    }
    __syncthreads();

    // stage 2: msg = (h @ mw2.T + mb2) * alpha, scatter-add into agg
    v16bf ah[4];
#pragma unroll
    for (int kk = 0; kk < 4; ++kk) ah[kk] = load_a_frag_lds(sH, kk * 32, DIM);
    v8f c2 = {};
#pragma unroll
    for (int kk = 0; kk < 4; ++kk)
      c2 = __builtin_amdgcn_wmma_f32_16x16x32_bf16(false, ah[kk], false, b2[kk],
                                                   (short)0, c2, false, false);
#pragma unroll
    for (int v = 0; v < 8; ++v) {
      int m = v + (half << 3);
      float d = (c2[v] + bias2) * sAlpha[m];
      atomicAdd(&agg[(size_t)sSrc[m] * DIM + nbase + col], d);  // L2-resident
    }
  }
}

// ---------------- residual + LayerNorm: one wave per node --------------------
__global__ __launch_bounds__(256) void layernorm_kernel(
    const float* __restrict__ z, const float* __restrict__ agg,
    const float* __restrict__ gamma, const float* __restrict__ beta,
    float* __restrict__ out, int n) {
  int row = blockIdx.x * 8 + (threadIdx.x >> 5);
  if (row >= n) return;
  int lane = threadIdx.x & 31;
  float x[4], s = 0.f;
#pragma unroll
  for (int i = 0; i < 4; ++i) {
    int d = lane + 32 * i;
    x[i] = z[(size_t)row * DIM + d] + agg[(size_t)row * DIM + d];
    s += x[i];
  }
#pragma unroll
  for (int off = 16; off > 0; off >>= 1) s += __shfl_xor(s, off, 32);
  float mu = s * (1.f / DIM);
  float var = 0.f;
#pragma unroll
  for (int i = 0; i < 4; ++i) { float t = x[i] - mu; var += t * t; }
#pragma unroll
  for (int off = 16; off > 0; off >>= 1) var += __shfl_xor(var, off, 32);
  float inv = rsqrtf(var * (1.f / DIM) + 1e-5f);
#pragma unroll
  for (int i = 0; i < 4; ++i) {
    int d = lane + 32 * i;
    out[(size_t)row * DIM + d] = (x[i] - mu) * inv * gamma[d] + beta[d];
  }
}

extern "C" void kernel_launch(void* const* d_in, const int* in_sizes, int n_in,
                              void* d_out, int out_size, void* d_ws,
                              size_t ws_size, hipStream_t stream) {
  (void)in_sizes; (void)n_in; (void)out_size; (void)ws_size;
  const float* z        = (const float*)d_in[0];
  const int*   ei       = (const int*)d_in[1];
  const int*   etype    = (const int*)d_in[2];
  const float* type_emb = (const float*)d_in[3];
  const float* ew1      = (const float*)d_in[4];
  const float* eb1      = (const float*)d_in[5];
  const float* ew2      = (const float*)d_in[6];
  const float* eb2      = (const float*)d_in[7];
  const float* mw1      = (const float*)d_in[8];
  const float* mb1      = (const float*)d_in[9];
  const float* mw2      = (const float*)d_in[10];
  const float* mb2      = (const float*)d_in[11];
  const float* gamma    = (const float*)d_in[12];
  const float* beta     = (const float*)d_in[13];

  const int* src = ei;
  const int* dst = ei + EE;

  char* ws = (char*)d_ws;
  auto au = [](size_t x) { return (x + 255) & ~(size_t)255; };
  size_t o_ew1 = 0;
  size_t o_mw1 = au(o_ew1 + (size_t)DIM * INP * 2);
  size_t o_mw2 = au(o_mw1 + (size_t)DIM * DIM * 2);
  size_t o_w   = au(o_mw2 + (size_t)DIM * DIM * 2);
  size_t o_we  = au(o_w   + (size_t)EE * 4);
  size_t o_max = au(o_we  + (size_t)EE * 4);
  size_t o_sum = au(o_max + (size_t)NN * 4);
  size_t o_agg = au(o_sum + (size_t)NN * 4);

  unsigned short* ew1bf = (unsigned short*)(ws + o_ew1);
  unsigned short* mw1bf = (unsigned short*)(ws + o_mw1);
  unsigned short* mw2bf = (unsigned short*)(ws + o_mw2);
  float* wbuf   = (float*)(ws + o_w);
  float* wexpb  = (float*)(ws + o_we);
  float* maxbuf = (float*)(ws + o_max);
  float* sumbuf = (float*)(ws + o_sum);
  float* agg    = (float*)(ws + o_agg);

  float* out       = (float*)d_out;
  float* alpha_out = out + (size_t)NN * DIM;

  const int numTiles = EE / 16;

  prep_weights_kernel<<<(DIM * INP + 255) / 256, 256, 0, stream>>>(
      ew1, mw1, mw2, ew1bf, mw1bf, mw2bf);
  init_kernel<<<(NN * DIM + 255) / 256, 256, 0, stream>>>(agg, maxbuf, sumbuf);
  edge_mlp_kernel<<<2048, 256, 0, stream>>>(z, src, dst, etype, type_emb, ew1bf,
                                            eb1, ew2, eb2, wbuf,
                                            (unsigned int*)maxbuf, numTiles);
  softmax_exp_kernel<<<(EE + 255) / 256, 256, 0, stream>>>(wbuf, src, maxbuf,
                                                           wexpb, sumbuf, EE);
  message_kernel<<<2048, 256, 0, stream>>>(z, src, dst, mw1bf, mb1, mw2bf, mb2,
                                           wexpb, sumbuf, alpha_out, agg,
                                           numTiles);
  layernorm_kernel<<<(NN + 7) / 8, 256, 0, stream>>>(z, agg, gamma, beta, out,
                                                     NN);
}
